// LocalEncoder_29575144800261
// MI455X (gfx1250) — compile-verified, matrix operand order checked by reading, not run on hardware
//
#include <hip/hip_runtime.h>
#include <hip/hip_bf16.h>

// ---------------------------------------------------------------------------
// HiVT LocalEncoder for MI455X (gfx1250, wave32, WMMA f32_16x16x32_f16).
// Input flattening assumption: setup_inputs() dict insertion order, params
// nested-dict leaves depth-first in insertion order.
// ---------------------------------------------------------------------------

typedef __attribute__((ext_vector_type(16))) _Float16 v16h;
typedef __attribute__((ext_vector_type(8)))  float    v8f;
typedef _Float16 h16;

#define TT 20
#define NN 1000
#define TN 20000     // T*N rows
#define EE 320000
#define DD 64

union FragH { v16h v; uint4 q[2]; h16 h[16]; };

// ---- fragment loaders -----------------------------------------------------
// A-matrix 16x32 f16 layout (ISA 7.12.2): lane (hf = l>>4, m = l&15):
//   elems 0..7  = X[m][kb + hf*8 + 0..7]
//   elems 8..15 = X[m][kb + 16 + hf*8 + 0..7]
// B-matrix 32x16 assumed symmetric with N in place of M. C/D: row r+8*hf, col n.
__device__ __forceinline__ v16h ld_h16(const h16* P, int ldk, int row, int kb, int hf) {
    FragH f;
    const h16* p = P + (size_t)row * ldk + kb + hf * 8;
    f.q[0] = *(const uint4*)p;
    f.q[1] = *(const uint4*)(p + 16);
    return f.v;
}

__device__ __forceinline__ v16h ld_f32_as_h(const float* P, int ldk, int row, int kb, int hf) {
    const float* p = P + (size_t)row * ldk + kb + hf * 8;
    FragH f;
#pragma unroll
    for (int i = 0; i < 8; ++i) { f.h[i] = (h16)p[i]; f.h[8 + i] = (h16)p[16 + i]; }
    return f.v;
}

// ---- 16x64 GEMM: out[16][64] = A(16x64) @ W^T + bias  (W row-major [64][ldk])
__device__ __forceinline__ void gemm64(v16h a0, v16h a1, const h16* W, int ldk,
                                       const float* bias, int n, int hf, v8f out[4]) {
#pragma unroll
    for (int cc = 0; cc < 4; ++cc) {
        int col = n + 16 * cc;
        float bv = bias[col];
        v8f acc;
#pragma unroll
        for (int r = 0; r < 8; ++r) acc[r] = bv;
        v16h b0 = ld_h16(W, ldk, col, 0, hf);
        v16h b1 = ld_h16(W, ldk, col, 32, hf);
        acc = __builtin_amdgcn_wmma_f32_16x16x32_f16(false, a0, false, b0, (short)0, acc, false, false);
        acc = __builtin_amdgcn_wmma_f32_16x16x32_f16(false, a1, false, b1, (short)0, acc, false, false);
        out[cc] = acc;
    }
}

// ---- LayerNorm over 64 features held in C-layout accumulators --------------
__device__ __forceinline__ void layernorm(v8f c[4], const float* g, const float* b,
                                          int n, bool relu) {
    float gv[4], bv[4];
#pragma unroll
    for (int cc = 0; cc < 4; ++cc) { gv[cc] = g[n + 16 * cc]; bv[cc] = b[n + 16 * cc]; }
#pragma unroll
    for (int r = 0; r < 8; ++r) {
        float s = 0.f, q = 0.f;
#pragma unroll
        for (int cc = 0; cc < 4; ++cc) { float x = c[cc][r]; s += x; q += x * x; }
#pragma unroll
        for (int mk = 1; mk < 16; mk <<= 1) { s += __shfl_xor(s, mk, 32); q += __shfl_xor(q, mk, 32); }
        float mean = s * (1.f / 64.f);
        float var  = q * (1.f / 64.f) - mean * mean;
        float rstd = rsqrtf(var + 1e-5f);
#pragma unroll
        for (int cc = 0; cc < 4; ++cc) {
            float y = (c[cc][r] - mean) * rstd * gv[cc] + bv[cc];
            if (relu) y = fmaxf(y, 0.f);
            c[cc][r] = y;
        }
    }
}

__device__ __forceinline__ void stage16(const v8f c[4], h16* S, int ld, int n, int hf) {
#pragma unroll
    for (int cc = 0; cc < 4; ++cc)
#pragma unroll
        for (int r = 0; r < 8; ++r)
            S[(r + 8 * hf) * ld + n + 16 * cc] = (h16)c[cc][r];
}

// 2->64 linear directly in C-layout (per-row scalars f0,f1 in LDS)
__device__ __forceinline__ void lin2to64(const float* f0, const float* f1, const float* w,
                                         const float* b, int n, int hf, v8f c[4]) {
#pragma unroll
    for (int cc = 0; cc < 4; ++cc) {
        int col = n + 16 * cc;
        float w0 = w[col * 2 + 0], w1 = w[col * 2 + 1], bb = b[col];
#pragma unroll
        for (int r = 0; r < 8; ++r) { int row = r + 8 * hf; c[cc][r] = f0[row] * w0 + f1[row] * w1 + bb; }
    }
}

// ordered-uint encoding for float atomicMax
__device__ __forceinline__ unsigned fenc(float f) {
    unsigned b = __float_as_uint(f);
    return (b & 0x80000000u) ? ~b : (b | 0x80000000u);
}
__device__ __forceinline__ float fdec(unsigned u) {
    unsigned b = (u & 0x80000000u) ? (u & 0x7fffffffu) : ~u;
    return __uint_as_float(b);
}

// ---------------------------------------------------------------------------
// Kernel 0: weight f32->f16 conversion + accumulator re-init (every launch)
// ---------------------------------------------------------------------------
struct PrepArgs {
    const float* src[14];          // 12 x [64x64], mlp1 [256x64], mlp2 [64x256]
    h16* wdst; float* agg; float* den; unsigned* amaxu;
};
__global__ __launch_bounds__(256) void k_prep(PrepArgs a) {
    const long total = 81920L + (long)TN * 64 + (long)TN * 8 * 2;
    long stride = (long)gridDim.x * blockDim.x;
    for (long i = (long)blockIdx.x * blockDim.x + threadIdx.x; i < total; i += stride) {
        if (i < 49152)       { int w = (int)(i >> 12); a.wdst[i] = (h16)a.src[w][i & 4095]; }
        else if (i < 65536)  { a.wdst[i] = (h16)a.src[12][i - 49152]; }
        else if (i < 81920)  { a.wdst[i] = (h16)a.src[13][i - 65536]; }
        else {
            long j = i - 81920;
            if (j < (long)TN * 64)      a.agg[j] = 0.f;
            else { j -= (long)TN * 64;
                   if (j < TN * 8)      a.den[j] = 0.f;
                   else                 a.amaxu[j - TN * 8] = 0u; }
        }
    }
}

// ---------------------------------------------------------------------------
// Kernel 1: center embedding + bos + norm1  -> ce (f32) and ce16 (f16)
// ---------------------------------------------------------------------------
struct K1A {
    const float *x, *rot, *bos_tok;
    const unsigned char* bos;
    const float *l1w, *l1b, *n1g, *n1b, *l2b, *n2g, *n2b, *l3b, *n3g, *n3b, *nrm1g, *nrm1b;
    const h16 *W2, *W3;
    float* ce; h16* ce16;
};
__global__ __launch_bounds__(320) void k_center(K1A A) {
    __shared__ __align__(16) h16 sA[10][16 * 72];
    __shared__ float sx0[10][16], sx1[10][16];
    int lane = threadIdx.x & 31, wv = threadIdx.x >> 5;
    int tile = blockIdx.x * 10 + wv;
    int rb = tile * 16;
    int n = lane & 15, hf = lane >> 4, m = n;
    if (lane < 16) {
        int g = rb + lane;
        int node = g % NN;
        const float* R = A.rot + node * 4;   // R[i*2+j]
        float x0 = A.x[(size_t)g * 2], x1 = A.x[(size_t)g * 2 + 1];
        sx0[wv][lane] = x0 * R[0] + x1 * R[2];
        sx1[wv][lane] = x0 * R[1] + x1 * R[3];
    }
    __builtin_amdgcn_wave_barrier();
    v8f c[4];
    lin2to64(sx0[wv], sx1[wv], A.l1w, A.l1b, n, hf, c);
    layernorm(c, A.n1g, A.n1b, n, true);
    stage16(c, sA[wv], 72, n, hf);
    __builtin_amdgcn_wave_barrier();
    { v16h a0 = ld_h16(sA[wv], 72, m, 0, hf), a1 = ld_h16(sA[wv], 72, m, 32, hf);
      gemm64(a0, a1, A.W2, 64, A.l2b, n, hf, c); }
    layernorm(c, A.n2g, A.n2b, n, true);
    __builtin_amdgcn_wave_barrier();
    stage16(c, sA[wv], 72, n, hf);
    __builtin_amdgcn_wave_barrier();
    { v16h a0 = ld_h16(sA[wv], 72, m, 0, hf), a1 = ld_h16(sA[wv], 72, m, 32, hf);
      gemm64(a0, a1, A.W3, 64, A.l3b, n, hf, c); }
    layernorm(c, A.n3g, A.n3b, n, false);
    // bos substitution (mask [N,T], token [T,64])
#pragma unroll
    for (int cc = 0; cc < 4; ++cc) {
        int col = n + 16 * cc;
#pragma unroll
        for (int r = 0; r < 8; ++r) {
            int g = rb + r + 8 * hf;
            int t = g / NN, node = g % NN;
            if (A.bos[node * TT + t]) c[cc][r] = A.bos_tok[t * DD + col];
        }
    }
    layernorm(c, A.nrm1g, A.nrm1b, n, false);
#pragma unroll
    for (int cc = 0; cc < 4; ++cc) {
        int col = n + 16 * cc;
#pragma unroll
        for (int r = 0; r < 8; ++r) {
            size_t g = rb + r + 8 * hf;
            A.ce[g * DD + col]   = c[cc][r];
            A.ce16[g * DD + col] = (h16)c[cc][r];
        }
    }
}

// ---------------------------------------------------------------------------
// Kernel 2: nbr embedding, q/k/v, attention logits + segment max
// ---------------------------------------------------------------------------
struct K2A {
    const float *x, *rot, *ea;
    const long long *src, *dst;
    const float *i0l1w, *i0l1b, *i0n1g, *i0n1b, *i0l2b;
    const float *i1l1w, *i1l1b, *i1n1g, *i1n1b, *i1l2b;
    const float *agn1g, *agn1b, *aglb, *agn2g, *agn2b;
    const float *qb, *kb, *vb;
    const h16 *Wi0, *Wi1, *Wag, *Wq, *Wk, *Wv;
    const h16* ce16;
    h16* vbuf; float* logit; unsigned* amaxu;
};
__global__ __launch_bounds__(320) void k_edge(K2A A) {
    __shared__ __align__(16) h16 sA[10][16 * 72];
    __shared__ float sf0[10][16], sf1[10][16], se0[10][16], se1[10][16];
    __shared__ int sd[10][16];
    int lane = threadIdx.x & 31, wv = threadIdx.x >> 5;
    int tile = blockIdx.x * 10 + wv;
    int eb = tile * 16;
    int n = lane & 15, hf = lane >> 4, m = n;
    if (lane < 16) {
        int e = eb + lane;
        long long s = A.src[e], d = A.dst[e];
        int node = (int)(d % NN);
        const float* R = A.rot + node * 4;
        float x0 = A.x[(size_t)s * 2], x1 = A.x[(size_t)s * 2 + 1];
        float e0 = A.ea[(size_t)e * 2], e1 = A.ea[(size_t)e * 2 + 1];
        sf0[wv][lane] = x0 * R[0] + x1 * R[2];
        sf1[wv][lane] = x0 * R[1] + x1 * R[3];
        se0[wv][lane] = e0 * R[0] + e1 * R[2];
        se1[wv][lane] = e0 * R[1] + e1 * R[3];
        sd[wv][lane] = (int)d;
    }
    __builtin_amdgcn_wave_barrier();
    v8f c[4], sum[4];
    // branch 0 (rotated x_j)
    lin2to64(sf0[wv], sf1[wv], A.i0l1w, A.i0l1b, n, hf, c);
    layernorm(c, A.i0n1g, A.i0n1b, n, true);
    stage16(c, sA[wv], 72, n, hf);
    __builtin_amdgcn_wave_barrier();
    { v16h a0 = ld_h16(sA[wv], 72, m, 0, hf), a1 = ld_h16(sA[wv], 72, m, 32, hf);
      gemm64(a0, a1, A.Wi0, 64, A.i0l2b, n, hf, sum); }
    // branch 1 (rotated edge_attr)
    lin2to64(se0[wv], se1[wv], A.i1l1w, A.i1l1b, n, hf, c);
    layernorm(c, A.i1n1g, A.i1n1b, n, true);
    __builtin_amdgcn_wave_barrier();
    stage16(c, sA[wv], 72, n, hf);
    __builtin_amdgcn_wave_barrier();
    { v16h a0 = ld_h16(sA[wv], 72, m, 0, hf), a1 = ld_h16(sA[wv], 72, m, 32, hf);
      gemm64(a0, a1, A.Wi1, 64, A.i1l2b, n, hf, c); }
#pragma unroll
    for (int cc = 0; cc < 4; ++cc) sum[cc] += c[cc];
    // aggregate MLP
    layernorm(sum, A.agn1g, A.agn1b, n, true);
    __builtin_amdgcn_wave_barrier();
    stage16(sum, sA[wv], 72, n, hf);
    __builtin_amdgcn_wave_barrier();
    { v16h a0 = ld_h16(sA[wv], 72, m, 0, hf), a1 = ld_h16(sA[wv], 72, m, 32, hf);
      gemm64(a0, a1, A.Wag, 64, A.aglb, n, hf, c); }
    layernorm(c, A.agn2g, A.agn2b, n, false);     // c = nbr embedding
    __builtin_amdgcn_wave_barrier();
    stage16(c, sA[wv], 72, n, hf);
    __builtin_amdgcn_wave_barrier();
    v8f kac[4], vac[4], qac[4];
    { v16h a0 = ld_h16(sA[wv], 72, m, 0, hf), a1 = ld_h16(sA[wv], 72, m, 32, hf);
      gemm64(a0, a1, A.Wk, 64, A.kb, n, hf, kac);
      gemm64(a0, a1, A.Wv, 64, A.vb, n, hf, vac); }
    { int drow = sd[wv][m];
      v16h a0 = ld_h16(A.ce16, 64, drow, 0, hf), a1 = ld_h16(A.ce16, 64, drow, 32, hf);
      gemm64(a0, a1, A.Wq, 64, A.qb, n, hf, qac); }
    // store v (f16) per edge
#pragma unroll
    for (int cc = 0; cc < 4; ++cc) {
        int col = n + 16 * cc;
#pragma unroll
        for (int r = 0; r < 8; ++r) {
            size_t e = (size_t)eb + r + 8 * hf;
            A.vbuf[e * DD + col] = (h16)vac[cc][r];
        }
    }
    // per-head logits: head = col/8 = 2*cc + (n>=8); reduce over 8 lanes
#pragma unroll
    for (int cc = 0; cc < 4; ++cc) {
#pragma unroll
        for (int r = 0; r < 8; ++r) {
            float p = qac[cc][r] * kac[cc][r];
            p += __shfl_xor(p, 1, 32);
            p += __shfl_xor(p, 2, 32);
            p += __shfl_xor(p, 4, 32);
            p *= 0.35355339059327373f;            // 1/sqrt(DH=8)
            if ((n & 7) == 0) {
                int head = 2 * cc + (n >> 3);
                int row = r + 8 * hf;
                int e = eb + row, d = sd[wv][row];
                A.logit[(size_t)e * 8 + head] = p;
                atomicMax(&A.amaxu[d * 8 + head], fenc(p));
            }
        }
    }
}

// ---------------------------------------------------------------------------
// Kernel 2b: exp(logit - max), segment denominator
// ---------------------------------------------------------------------------
__global__ __launch_bounds__(256) void k_softmax_den(const long long* dst, const unsigned* amaxu,
                                                     float* exbuf, float* den) {
    long i = (long)blockIdx.x * blockDim.x + threadIdx.x;   // exactly E*8
    long e = i >> 3; int h = (int)(i & 7);
    int d = (int)dst[e];
    float ex = __expf(exbuf[i] - fdec(amaxu[d * 8 + h]));
    exbuf[i] = ex;
    atomicAdd(&den[d * 8 + h], ex);
}

// ---------------------------------------------------------------------------
// Kernel 2c: agg[d] += v * alpha
// ---------------------------------------------------------------------------
__global__ __launch_bounds__(256) void k_aggregate(const long long* dst, const float* exbuf,
                                                   const float* den, const h16* vbuf, float* agg) {
    long i = (long)blockIdx.x * blockDim.x + threadIdx.x;   // exactly E*64
    long e = i >> 6; int c = (int)(i & 63), h = c >> 3;
    int d = (int)dst[e];
    float w = exbuf[e * 8 + h] / (den[d * 8 + h] + 1e-16f);
    atomicAdd(&agg[(size_t)d * DD + c], (float)vbuf[i] * w);
}

// ---------------------------------------------------------------------------
// Kernel 3: gated update + out_proj + norm2 + MLP(64->256->64) + residuals
// ---------------------------------------------------------------------------
struct K3A {
    const float* ce; const h16* ce16; const float* agg;
    const float *ihb, *hhb, *selfb, *opb, *n2g, *n2b, *m1b, *m2b;
    const h16 *Wih, *Whh, *Wself, *Wop, *Wm1, *Wm2;
    float* out;
};
__global__ __launch_bounds__(128) void k_update(K3A A) {
    __shared__ __align__(16) h16 sA[4][16 * 264];
    int lane = threadIdx.x & 31, wv = threadIdx.x >> 5;
    int tile = blockIdx.x * 4 + wv;
    if (tile >= TN / 16) return;
    int rb = tile * 16;
    int n = lane & 15, hf = lane >> 4, m = n;
    h16* S = sA[wv];
    v8f gi[4], gh[4], sf[4];
    { v16h a0 = ld_f32_as_h(A.agg, 64, rb + m, 0, hf), a1 = ld_f32_as_h(A.agg, 64, rb + m, 32, hf);
      gemm64(a0, a1, A.Wih, 64, A.ihb, n, hf, gi); }
    { v16h a0 = ld_h16(A.ce16, 64, rb + m, 0, hf), a1 = ld_h16(A.ce16, 64, rb + m, 32, hf);
      gemm64(a0, a1, A.Whh, 64, A.hhb, n, hf, gh);
      gemm64(a0, a1, A.Wself, 64, A.selfb, n, hf, sf); }
    v8f upd[4];
#pragma unroll
    for (int cc = 0; cc < 4; ++cc) {
        int col = n + 16 * cc;
#pragma unroll
        for (int r = 0; r < 8; ++r) {
            float a = A.agg[(size_t)(rb + r + 8 * hf) * DD + col];
            float g = 1.f / (1.f + __expf(-(gi[cc][r] + gh[cc][r])));
            upd[cc][r] = a + g * (sf[cc][r] - a);
        }
    }
    stage16(upd, S, 264, n, hf);
    __builtin_amdgcn_wave_barrier();
    v8f oc[4];
    { v16h a0 = ld_h16(S, 264, m, 0, hf), a1 = ld_h16(S, 264, m, 32, hf);
      gemm64(a0, a1, A.Wop, 64, A.opb, n, hf, oc); }
    v8f ce2[4], hC[4];
#pragma unroll
    for (int cc = 0; cc < 4; ++cc) {
        int col = n + 16 * cc;
#pragma unroll
        for (int r = 0; r < 8; ++r)
            ce2[cc][r] = A.ce[(size_t)(rb + r + 8 * hf) * DD + col] + oc[cc][r];
        hC[cc] = ce2[cc];
    }
    layernorm(hC, A.n2g, A.n2b, n, false);
    __builtin_amdgcn_wave_barrier();
    stage16(hC, S, 264, n, hf);
    __builtin_amdgcn_wave_barrier();
    v16h h0 = ld_h16(S, 264, m, 0, hf), h1 = ld_h16(S, 264, m, 32, hf);
    __builtin_amdgcn_wave_barrier();
    // mlp1: 64 -> 256, relu, staged to LDS (stride 264)
#pragma unroll
    for (int ct = 0; ct < 16; ++ct) {
        int col = ct * 16 + n;
        float bv = A.m1b[col];
        v8f acc;
#pragma unroll
        for (int r = 0; r < 8; ++r) acc[r] = bv;
        v16h b0 = ld_h16(A.Wm1, 64, col, 0, hf);
        v16h b1 = ld_h16(A.Wm1, 64, col, 32, hf);
        acc = __builtin_amdgcn_wmma_f32_16x16x32_f16(false, h0, false, b0, (short)0, acc, false, false);
        acc = __builtin_amdgcn_wmma_f32_16x16x32_f16(false, h1, false, b1, (short)0, acc, false, false);
#pragma unroll
        for (int r = 0; r < 8; ++r) S[(r + 8 * hf) * 264 + col] = (h16)fmaxf(acc[r], 0.f);
    }
    __builtin_amdgcn_wave_barrier();
    // mlp2: 256 -> 64 (K = 256 in 8 WMMA steps per column tile)
    v16h av[8];
#pragma unroll
    for (int ks = 0; ks < 8; ++ks) av[ks] = ld_h16(S, 264, m, ks * 32, hf);
    v8f m2[4];
#pragma unroll
    for (int cc = 0; cc < 4; ++cc) {
        int col = n + 16 * cc;
        float bv = A.m2b[col];
        v8f acc;
#pragma unroll
        for (int r = 0; r < 8; ++r) acc[r] = bv;
#pragma unroll
        for (int ks = 0; ks < 8; ++ks) {
            v16h b = ld_h16(A.Wm2, 256, col, ks * 32, hf);
            acc = __builtin_amdgcn_wmma_f32_16x16x32_f16(false, av[ks], false, b, (short)0, acc, false, false);
        }
        m2[cc] = acc;
    }
#pragma unroll
    for (int cc = 0; cc < 4; ++cc) {
        int col = n + 16 * cc;
#pragma unroll
        for (int r = 0; r < 8; ++r)
            A.out[(size_t)(rb + r + 8 * hf) * DD + col] = ce2[cc][r] + m2[cc][r];
    }
}

// ---------------------------------------------------------------------------
extern "C" void kernel_launch(void* const* d_in, const int* in_sizes, int n_in,
                              void* d_out, int out_size, void* d_ws, size_t ws_size,
                              hipStream_t stream) {
    (void)in_sizes; (void)n_in; (void)out_size; (void)ws_size;
    auto F = [&](int i) { return (const float*)d_in[i]; };
    const float* x            = F(0);
    const long long* eidx     = (const long long*)d_in[1];
    const float* eattr        = F(2);
    const unsigned char* bos  = (const unsigned char*)d_in[3];
    const float* rot          = F(4);
    // params leaves (insertion-order DFS), base index 5 — see analysis
    const long long* esrc = eidx;
    const long long* edst = eidx + EE;

    char* ws = (char*)d_ws;
    h16*      wf16  = (h16*)ws;                          //   163,840 B
    float*    cebuf = (float*)(ws + 163840);             // 5,120,000 B
    h16*      ce16  = (h16*)(ws + 5283840);              // 2,560,000 B
    h16*      vbuf  = (h16*)(ws + 7843840);              // 40,960,000 B
    float*    exbuf = (float*)(ws + 48803840);           // 10,240,000 B
    unsigned* amaxu = (unsigned*)(ws + 59043840);        //   640,000 B
    float*    den   = (float*)(ws + 59683840);           //   640,000 B
    float*    agg   = (float*)(ws + 60323840);           // 5,120,000 B

    const h16* W_cel2 = wf16 +  0 * 4096;
    const h16* W_cel3 = wf16 +  1 * 4096;
    const h16* W_i0   = wf16 +  2 * 4096;
    const h16* W_i1   = wf16 +  3 * 4096;
    const h16* W_ag   = wf16 +  4 * 4096;
    const h16* W_q    = wf16 +  5 * 4096;
    const h16* W_k    = wf16 +  6 * 4096;
    const h16* W_v    = wf16 +  7 * 4096;
    const h16* W_self = wf16 +  8 * 4096;
    const h16* W_ih   = wf16 +  9 * 4096;
    const h16* W_hh   = wf16 + 10 * 4096;
    const h16* W_op   = wf16 + 11 * 4096;
    const h16* W_m1   = wf16 + 49152;
    const h16* W_m2   = wf16 + 65536;

    PrepArgs P;
    P.src[0] = F(9);  P.src[1] = F(13); P.src[2]  = F(21); P.src[3]  = F(27);
    P.src[4] = F(31); P.src[5] = F(35); P.src[6]  = F(37); P.src[7]  = F(39);
    P.src[8] = F(41); P.src[9] = F(43); P.src[10] = F(45); P.src[11] = F(47);
    P.src[12] = F(53); P.src[13] = F(55);
    P.wdst = wf16; P.agg = agg; P.den = den; P.amaxu = amaxu;
    k_prep<<<2048, 256, 0, stream>>>(P);

    K1A a1;
    a1.x = x; a1.rot = rot; a1.bos_tok = F(57); a1.bos = bos;
    a1.l1w = F(5);  a1.l1b = F(6);  a1.n1g = F(7);  a1.n1b = F(8);
    a1.l2b = F(10); a1.n2g = F(11); a1.n2b = F(12);
    a1.l3b = F(14); a1.n3g = F(15); a1.n3b = F(16);
    a1.nrm1g = F(49); a1.nrm1b = F(50);
    a1.W2 = W_cel2; a1.W3 = W_cel3;
    a1.ce = cebuf; a1.ce16 = ce16;
    k_center<<<125, 320, 0, stream>>>(a1);       // 1250 tiles of 16 rows

    K2A a2;
    a2.x = x; a2.rot = rot; a2.ea = eattr; a2.src = esrc; a2.dst = edst;
    a2.i0l1w = F(17); a2.i0l1b = F(18); a2.i0n1g = F(19); a2.i0n1b = F(20); a2.i0l2b = F(22);
    a2.i1l1w = F(23); a2.i1l1b = F(24); a2.i1n1g = F(25); a2.i1n1b = F(26); a2.i1l2b = F(28);
    a2.agn1g = F(29); a2.agn1b = F(30); a2.aglb = F(32); a2.agn2g = F(33); a2.agn2b = F(34);
    a2.qb = F(36); a2.kb = F(38); a2.vb = F(40);
    a2.Wi0 = W_i0; a2.Wi1 = W_i1; a2.Wag = W_ag; a2.Wq = W_q; a2.Wk = W_k; a2.Wv = W_v;
    a2.ce16 = ce16; a2.vbuf = vbuf; a2.logit = exbuf; a2.amaxu = amaxu;
    k_edge<<<2000, 320, 0, stream>>>(a2);        // 20000 tiles of 16 edges

    k_softmax_den<<<10000, 256, 0, stream>>>(edst, amaxu, exbuf, den);   // E*8
    k_aggregate<<<80000, 256, 0, stream>>>(edst, exbuf, den, vbuf, agg); // E*64

    K3A a3;
    a3.ce = cebuf; a3.ce16 = ce16; a3.agg = agg;
    a3.ihb = F(44); a3.hhb = F(46); a3.selfb = F(42); a3.opb = F(48);
    a3.n2g = F(51); a3.n2b = F(52); a3.m1b = F(54); a3.m2b = F(56);
    a3.Wih = W_ih; a3.Whh = W_hh; a3.Wself = W_self; a3.Wop = W_op;
    a3.Wm1 = W_m1; a3.Wm2 = W_m2;
    a3.out = (float*)d_out;
    k_update<<<313, 128, 0, stream>>>(a3);       // 1250 tiles, 4 waves/block
}
